// SSMLayer_30348238914010
// MI455X (gfx1250) — matrix operand order repaired and checked by reading, not compiled, not run on hardware
//
#include <hip/hip_runtime.h>

// ---------------------------------------------------------------------------
// SSM layer for MI455X (gfx1250, wave32, WMMA).
//   1) Bilinear discretization via Newton-Schulz inverse (f32 WMMA GEMMs)
//   2) K_l = C Abar^l Bbar via doubling (f32 WMMA GEMMs)
//   3) y = K * u causal conv via Toeplitz-block bf16 WMMA (f32 accumulate)
// ---------------------------------------------------------------------------

constexpr int NDIM = 256;       // latent dim
constexpr int NN   = NDIM * NDIM;
constexpr int LSEQ = 131072;    // sequence length
constexpr int MW   = 512;       // width of W   (K reshaped as 256 x 512)
constexpr int JR   = 256;       // height of R
constexpr int PADU = 288;       // front zero-pad of u (covers ia0 >= -272)
constexpr int PADK = 64;        // zero-pad of reversed K copies

typedef __attribute__((ext_vector_type(16))) __bf16   v16bf;
typedef __attribute__((ext_vector_type(8)))  float    v8f;
typedef __attribute__((ext_vector_type(2)))  float    v2f;
typedef __attribute__((ext_vector_type(4)))  unsigned v4u;

// ---------------------------------------------------------------------------
// Small setup kernels
// ---------------------------------------------------------------------------

// Mm = I - (delta/2) A ; P0 = I + (delta/2) A
__global__ void ssm_prep(const float* __restrict__ A, const float* __restrict__ ldt,
                         float* __restrict__ Mm, float* __restrict__ P0) {
  int i = blockIdx.x * 256 + threadIdx.x;
  if (i >= NN) return;
  float delta = __expf(ldt[0]);
  int r = i >> 8, c = i & 255;
  float a  = A[i] * (0.5f * delta);
  float id = (r == c) ? 1.0f : 0.0f;
  Mm[i] = id - a;
  P0[i] = id + a;
}

// scale = 1 / (||M||_1 * ||M||_inf)   (one workgroup)
__global__ void ssm_norms(const float* __restrict__ Mm, float* __restrict__ scale) {
  __shared__ float rs[256], cs[256];
  int i = threadIdx.x;
  float r = 0.f, c = 0.f;
  for (int j = 0; j < NDIM; ++j) {
    r += fabsf(Mm[i * NDIM + j]);
    c += fabsf(Mm[j * NDIM + i]);
  }
  rs[i] = r; cs[i] = c;
  __syncthreads();
  for (int s = 128; s > 0; s >>= 1) {
    if (i < s) { rs[i] = fmaxf(rs[i], rs[i + s]); cs[i] = fmaxf(cs[i], cs[i + s]); }
    __syncthreads();
  }
  if (i == 0) scale[0] = 1.0f / (rs[0] * cs[0]);
}

// X0 = scale * M^T
__global__ void ssm_initx(const float* __restrict__ Mm, const float* __restrict__ scale,
                          float* __restrict__ X) {
  int i = blockIdx.x * 256 + threadIdx.x;
  if (i >= NN) return;
  int r = i >> 8, c = i & 255;
  X[i] = Mm[c * NDIM + r] * scale[0];
}

// in place: T = 2I - T
__global__ void ssm_gfun(float* __restrict__ T) {
  int i = blockIdx.x * 256 + threadIdx.x;
  if (i >= NN) return;
  int r = i >> 8, c = i & 255;
  T[i] = ((r == c) ? 2.0f : 0.0f) - T[i];
}

// Bbar = Xinv @ (delta * B)   (one workgroup, one row per thread)
__global__ void ssm_bbar(const float* __restrict__ X, const float* __restrict__ B,
                         const float* __restrict__ ldt, float* __restrict__ Bbar) {
  int t = threadIdx.x;
  float delta = __expf(ldt[0]);
  float s = 0.f;
  for (int j = 0; j < NDIM; ++j) s += X[t * NDIM + j] * B[j];
  Bbar[t] = s * delta;
}

// W[:, i] = Abar^i Bbar for i = 0..15 (sequential matvecs in LDS, one WG)
__global__ void ssm_wboot(const float* __restrict__ Abar, const float* __restrict__ Bbar,
                          float* __restrict__ W) {
  __shared__ float x[NDIM];
  int t = threadIdx.x;
  x[t] = Bbar[t];
  __syncthreads();
  for (int i = 0; i < 16; ++i) {
    W[t * MW + i] = x[t];
    float nx = 0.f;
    for (int j = 0; j < NDIM; ++j) nx += Abar[t * NDIM + j] * x[j];
    __syncthreads();
    x[t] = nx;
    __syncthreads();
  }
}

// R[j, :] = C P^j for j = 0..15 (sequential vec-mats in LDS, one WG)
__global__ void ssm_rboot(const float* __restrict__ Cm, const float* __restrict__ P,
                          float* __restrict__ R) {
  __shared__ float x[NDIM];
  int t = threadIdx.x;
  x[t] = Cm[t];
  __syncthreads();
  for (int j = 0; j < 16; ++j) {
    R[j * NDIM + t] = x[t];
    float nx = 0.f;
    for (int jj = 0; jj < NDIM; ++jj) nx += x[jj] * P[jj * NDIM + t];
    __syncthreads();
    x[t] = nx;
    __syncthreads();
  }
}

// ---------------------------------------------------------------------------
// f32 WMMA GEMM: C = A @ B, one 16x16 tile per wave (block = 32 threads).
// Uses V_WMMA_F32_16X16X4_F32; VALU fallback if the builtin is absent.
// ---------------------------------------------------------------------------
__global__ void ssm_gemm(const float* __restrict__ A, int lda,
                         const float* __restrict__ B, int ldb,
                         float* __restrict__ C, int ldc,
                         int M, int Nc, int Kd) {
  int lane = threadIdx.x & 31;
  int ntn  = Nc >> 4;
  int tm   = (blockIdx.x / ntn) << 4;
  int tn   = (blockIdx.x % ntn) << 4;
  int r    = lane & 15;
  int half = lane >> 4;
  union { v8f v; float f[8]; } acc;
#pragma unroll
  for (int i = 0; i < 8; ++i) acc.f[i] = 0.f;

#if __has_builtin(__builtin_amdgcn_wmma_f32_16x16x4_f32)
  for (int k0 = 0; k0 < Kd; k0 += 4) {
    // A 16x4: lanes 0-15 K={k0,k0+1}, lanes 16-31 K={k0+2,k0+3}; row = lane%16
    v2f a = *(const v2f*)&A[(tm + r) * lda + k0 + 2 * half];
    // B 4x16: lanes 0-15 rows {k0,k0+1}, lanes 16-31 rows {k0+2,k0+3}; col = lane%16
    v2f b;
    b.x = B[(k0 + 2 * half)     * ldb + tn + r];
    b.y = B[(k0 + 2 * half + 1) * ldb + tn + r];
    acc.v = __builtin_amdgcn_wmma_f32_16x16x4_f32(false, a, false, b,
                                                  (short)0, acc.v, false, false);
  }
#else
  for (int k = 0; k < Kd; ++k) {
    float bv = B[k * ldb + tn + r];
#pragma unroll
    for (int v = 0; v < 8; ++v)
      acc.f[v] += A[(tm + v + 8 * half) * lda + k] * bv;
  }
#endif

  // D layout: VGPR v -> row v (lanes 0-15) / v+8 (lanes 16-31), col = lane%16
#pragma unroll
  for (int v = 0; v < 8; ++v)
    C[(tm + v + 8 * half) * ldc + tn + r] = acc.f[v];
}

// ---------------------------------------------------------------------------
// Convert: build zero-padded bf16 u, and two reversed bf16 copies of K
// (offset by one element so per-lane b32 loads are always 4B-aligned).
// ---------------------------------------------------------------------------
__global__ void ssm_convert(const float* __restrict__ u, const float* __restrict__ Kv,
                            __bf16* __restrict__ ubp, __bf16* __restrict__ krA,
                            __bf16* __restrict__ krB) {
  int i = blockIdx.x * 256 + threadIdx.x;
  if (i < PADU + LSEQ + 64) {
    int j = i - PADU;
    ubp[i] = (j >= 0 && j < LSEQ) ? (__bf16)u[j] : (__bf16)0.0f;
  }
  if (i < PADK + LSEQ + 64) {
    int j = i - PADK;
    krA[i] = (j >= 0 && j < LSEQ)     ? (__bf16)Kv[LSEQ - 1 - j] : (__bf16)0.0f;
    krB[i] = (j >= 0 && j < LSEQ - 1) ? (__bf16)Kv[LSEQ - 2 - j] : (__bf16)0.0f;
  }
}

// ---------------------------------------------------------------------------
// Causal convolution y[t] = sum_j K[j] u[t-j] via bf16 WMMA.
// Per wave: 256 outputs (D[m][n] = y[t0+16m+n]); per K-chunk of 32:
//   A[m][k] = u[t0+16m - l0 - 16 + k]          (zero-padded, aligned b128)
//   B[k][n] = K[l0+16 - k + n]                 (reversed-K, aligned b32)
// Zero padding of u enforces causality; zero padding of K handles edges.
// ---------------------------------------------------------------------------
__global__ void ssm_conv(const __bf16* __restrict__ ubp, const __bf16* __restrict__ krA,
                         const __bf16* __restrict__ krB, float* __restrict__ y) {
  int tid  = threadIdx.x;
  int lane = tid & 31;
  int wave = tid >> 5;
  int gw   = blockIdx.x * (blockDim.x >> 5) + wave;
  int t0   = gw * 256;
  int n    = lane & 15;     // A row m and B col n
  int half = lane >> 4;

  union { v8f v; float f[8]; } acc;
#pragma unroll
  for (int i = 0; i < 8; ++i) acc.f[i] = 0.f;

  const __bf16* uA  = ubp + PADU;   // logical index 0 == u[0]
  const __bf16* kA  = krA + PADK;   // kA[i] = K[L-1-i]
  const __bf16* kB  = krB + PADK;   // kB[i] = K[L-2-i]

  int lmax = t0 + 288;
  for (int l0 = 0; l0 < lmax; l0 += 32) {
    // ---- A fragment (contiguous k order, same as B): per-lane k = 16*half + h, h = 0..15
    union { v16bf v; v4u q[2]; } af;
    int ia0 = t0 + 16 * n - l0 - 16 + 16 * half;  // multiple of 8 -> 16B aligned
    af.q[0] = *(const v4u*)(uA + ia0);
    af.q[1] = *(const v4u*)(uA + ia0 + 8);
    __builtin_prefetch((const void*)(uA + ia0 - 32), 0, 3);

    // ---- B fragment (32x16): per-lane k = 16*half + h, h = 0..15
    // B[k][n] = K[l0+16-k+n] = revK[s0 + (k - 16*half)], s0 below.
    union { v16bf v; unsigned w[8]; } bfr;
    int s0 = LSEQ - 17 - l0 + 16 * half - n;
    const __bf16* bb = (s0 & 1) ? (kB + (s0 - 1)) : (kA + s0);  // even base
#pragma unroll
    for (int i = 0; i < 8; ++i) bfr.w[i] = *(const unsigned*)(bb + 2 * i);

    acc.v = __builtin_amdgcn_wmma_f32_16x16x32_bf16(false, af.v, false, bfr.v,
                                                    (short)0, acc.v, false, false);
  }

#pragma unroll
  for (int v = 0; v < 8; ++v) {
    int row = v + 8 * half;
    y[t0 + 16 * row + n] = acc.f[v];
  }
}

// ---------------------------------------------------------------------------
// Host orchestration
// ---------------------------------------------------------------------------
extern "C" void kernel_launch(void* const* d_in, const int* in_sizes, int n_in,
                              void* d_out, int out_size, void* d_ws, size_t ws_size,
                              hipStream_t stream) {
  const float* u   = (const float*)d_in[0];
  const float* A   = (const float*)d_in[1];
  const float* B   = (const float*)d_in[2];
  const float* Cm  = (const float*)d_in[3];
  const float* ldt = (const float*)d_in[4];
  float* y = (float*)d_out;

  // ---- carve workspace (256B aligned slots)
  char* p = (char*)d_ws;
  auto carve = [&](size_t bytes) -> char* {
    char* r = p;
    p += (bytes + 255) & ~(size_t)255;
    return r;
  };
  float* Mm   = (float*)carve(NN * 4);
  float* P0   = (float*)carve(NN * 4);
  float* X    = (float*)carve(NN * 4);
  float* Xn   = (float*)carve(NN * 4);
  float* T1   = (float*)carve(NN * 4);
  float* Abar = (float*)carve(NN * 4);
  float* E    = (float*)carve(NN * 4);
  float* Et   = (float*)carve(NN * 4);
  float* Bbar = (float*)carve(NDIM * 4);
  float* W    = (float*)carve(NDIM * MW * 4);
  float* R    = (float*)carve(JR * NDIM * 4);
  float* Kv   = (float*)carve((size_t)LSEQ * 4);
  float* scl  = (float*)carve(256);
  __bf16* ubp = (__bf16*)carve((size_t)(PADU + LSEQ + 64) * 2);
  __bf16* krA = (__bf16*)carve((size_t)(PADK + LSEQ + 64) * 2);
  __bf16* krB = (__bf16*)carve((size_t)(PADK + LSEQ + 64) * 2);

  auto gemm = [&](float* Cp, int ldc, const float* Ap, int lda,
                  const float* Bp, int ldb, int M, int Nc, int Kd) {
    ssm_gemm<<<(M / 16) * (Nc / 16), 32, 0, stream>>>(Ap, lda, Bp, ldb, Cp, ldc, M, Nc, Kd);
  };

  // 1) discretization matrices + Newton-Schulz inverse of Mm
  ssm_prep<<<(NN + 255) / 256, 256, 0, stream>>>(A, ldt, Mm, P0);
  ssm_norms<<<1, 256, 0, stream>>>(Mm, scl);
  ssm_initx<<<(NN + 255) / 256, 256, 0, stream>>>(Mm, scl, X);
  float* Xc = X; float* Xo = Xn;
  for (int it = 0; it < 32; ++it) {
    gemm(T1, NDIM, Mm, NDIM, Xc, NDIM, NDIM, NDIM, NDIM);   // T1 = M @ X
    ssm_gfun<<<(NN + 255) / 256, 256, 0, stream>>>(T1);     // T1 = 2I - T1
    gemm(Xo, NDIM, Xc, NDIM, T1, NDIM, NDIM, NDIM, NDIM);   // X' = X @ (2I - MX)
    float* t = Xc; Xc = Xo; Xo = t;
  }

  // 2) Abar = X @ P0 ; Bbar = X @ (delta B)
  gemm(Abar, NDIM, Xc, NDIM, P0, NDIM, NDIM, NDIM, NDIM);
  ssm_bbar<<<1, 256, 0, stream>>>(Xc, B, ldt, Bbar);

  // 3) W = [Bbar, Abar Bbar, ..., Abar^511 Bbar] (N x 512) by doubling
  ssm_wboot<<<1, 256, 0, stream>>>(Abar, Bbar, W);          // cols 0..15
  hipMemcpyAsync(E, Abar, (size_t)NN * 4, hipMemcpyDeviceToDevice, stream);
  float* Ec = E; float* Eo = Et;
  for (int i = 0; i < 4; ++i) {                             // E = Abar^16
    gemm(Eo, NDIM, Ec, NDIM, Ec, NDIM, NDIM, NDIM, NDIM);
    float* t = Ec; Ec = Eo; Eo = t;
  }
  for (int k = 16; k <= 256; k <<= 1) {
    gemm(W + k, MW, Ec, NDIM, W, MW, NDIM, k, NDIM);        // cols [k,2k) = E @ cols [0,k)
    gemm(Eo, NDIM, Ec, NDIM, Ec, NDIM, NDIM, NDIM, NDIM);   // E = E^2
    float* t = Ec; Ec = Eo; Eo = t;
  }
  // Ec == Abar^512 == P

  // 4) R = [C; C P; ...; C P^255] (256 x N) by doubling
  ssm_rboot<<<1, 256, 0, stream>>>(Cm, Ec, R);              // rows 0..15
  for (int i = 0; i < 4; ++i) {                             // Q = P^16 (reuse E buffers)
    gemm(Eo, NDIM, Ec, NDIM, Ec, NDIM, NDIM, NDIM, NDIM);
    float* t = Ec; Ec = Eo; Eo = t;
  }
  for (int k = 16; k <= 128; k <<= 1) {
    gemm(R + k * NDIM, NDIM, R, NDIM, Ec, NDIM, k, NDIM, NDIM); // rows [k,2k)
    if (k < 128) {
      gemm(Eo, NDIM, Ec, NDIM, Ec, NDIM, NDIM, NDIM, NDIM);     // Q = Q^2
      float* t = Ec; Ec = Eo; Eo = t;
    }
  }

  // 5) K (256 x 512 row-major == flat length L) = R @ W
  gemm(Kv, MW, R, NDIM, W, MW, JR, MW, NDIM);

  // 6) bf16 conversions with zero padding, then WMMA convolution
  ssm_convert<<<(PADU + LSEQ + 64 + 255) / 256, 256, 0, stream>>>(u, Kv, ubp, krA, krB);
  ssm_conv<<<LSEQ / 2048, 256, 0, stream>>>(ubp, krA, krB, y);

  (void)in_sizes; (void)n_in; (void)out_size; (void)ws_size;
}
